// ProductionVectorQuantizer_82257213653156
// MI455X (gfx1250) — matrix operand order repaired and checked by reading, not compiled, not run on hardware
//
#include <hip/hip_runtime.h>

typedef __attribute__((ext_vector_type(16))) __bf16 v16bf;
typedef __attribute__((ext_vector_type(8)))  float  v8f;

typedef unsigned int   u32;
typedef unsigned short u16;
typedef __attribute__((ext_vector_type(4))) u32 u32x4;   // plain 16B vector

// explicit LDS (address_space 3) pointer => guaranteed ds_load_b128
typedef __attribute__((address_space(3))) const u32x4* lds_cu4p;
__device__ __forceinline__ u32x4 lds_read16(u32 byte_off) {
    return *(lds_cu4p)(size_t)byte_off;
}

#define D_DIM   1024
#define B_SZ    16
#define T_SZ    1500
#define M_TOT   24000          // B*T
#define M_PAD   24064          // padded to multiple of 128
#define NQ      8
#define KSTAGE0 2048
#define KREST   1024
#define NCB     (KSTAGE0 + 7*KREST)   // 9216 codebook rows total
#define OUT_ELEMS  ((size_t)B_SZ*D_DIM*T_SZ)   // 24,576,000
#define CODE_ELEMS ((size_t)B_SZ*NQ*T_SZ)      // 192,000

__device__ __forceinline__ u16 f2bf(float f) {
    u32 u = __float_as_uint(f);
    u32 r = (u + 0x7FFFu + ((u >> 16) & 1u)) >> 16;
    return (u16)r;
}

// async direct global->LDS copy, 16B per lane, tracked by ASYNCcnt.
// INST_OFFSET is applied to BOTH the global address and the LDS destination
// (ISA 08_async_tensor 4.4), so intra-tile sub-offsets ride the immediate.
__device__ __forceinline__ void async_b128(u32 lds_off, const void* gaddr, int ioff) {
    asm volatile("global_load_async_to_lds_b128 %0, %1, off offset:%2"
                 :: "v"(lds_off), "v"(gaddr), "n"(ioff)
                 : "memory");
}

// ---------------------------------------------------------------- zero ssq
__global__ void k_zero(float* ssq) {
    if (threadIdx.x < NQ) ssq[threadIdx.x] = 0.0f;
}

// ------------------------------------------- codebooks -> bf16 + ||c||^2
__global__ __launch_bounds__(256) void k_prep_cb(const float* __restrict__ cb0,
                                                 const float* __restrict__ cbr,
                                                 u16* __restrict__ cbB,
                                                 float* __restrict__ c2) {
    int row = blockIdx.x;
    const float* src = (row < KSTAGE0) ? cb0 + (size_t)row * D_DIM
                                       : cbr + (size_t)(row - KSTAGE0) * D_DIM;
    int tid = threadIdx.x;
    float4 v = ((const float4*)src)[tid];
    float s = v.x*v.x + v.y*v.y + v.z*v.z + v.w*v.w;
    uint2 p;
    p.x = (u32)f2bf(v.x) | ((u32)f2bf(v.y) << 16);
    p.y = (u32)f2bf(v.z) | ((u32)f2bf(v.w) << 16);
    ((uint2*)(cbB + (size_t)row * D_DIM))[tid] = p;
    #pragma unroll
    for (int m = 16; m >= 1; m >>= 1) s += __shfl_xor(s, m, 32);
    __shared__ float part[8];
    int lane = tid & 31, wave = tid >> 5;
    if (lane == 0) part[wave] = s;
    __syncthreads();
    if (tid == 0) {
        float t = 0.f;
        #pragma unroll
        for (int i = 0; i < 8; i++) t += part[i];
        c2[row] = t;
    }
}

// ----------------------------- z[B,D,T] -> residual[M,D] (f32 + bf16 shadow)
__global__ __launch_bounds__(256) void k_transpose(const float* __restrict__ z,
                                                   float* __restrict__ resF,
                                                   u16* __restrict__ resB) {
    __shared__ float tile[32][33];
    int b = blockIdx.x, t0 = blockIdx.y * 32, d0 = blockIdx.z * 32;
    int tid = threadIdx.x;
    int i  = tid >> 3;            // d-local row 0..31
    int j4 = (tid & 7) * 4;       // t-local col
    const float* zp = z + ((size_t)b * D_DIM + d0 + i) * T_SZ + t0;
    #pragma unroll
    for (int e = 0; e < 4; e++) {
        int t = t0 + j4 + e;
        tile[i][j4 + e] = (t < T_SZ) ? zp[j4 + e] : 0.0f;
    }
    __syncthreads();
    int ii  = tid >> 3;           // t-local row
    int jj4 = (tid & 7) * 4;      // d-local col
    int t = t0 + ii;
    if (t < T_SZ) {
        size_t m = (size_t)b * T_SZ + t;
        float4 v;
        v.x = tile[jj4 + 0][ii];
        v.y = tile[jj4 + 1][ii];
        v.z = tile[jj4 + 2][ii];
        v.w = tile[jj4 + 3][ii];
        *(float4*)(resF + m * D_DIM + d0 + jj4) = v;
        uint2 p;
        p.x = (u32)f2bf(v.x) | ((u32)f2bf(v.y) << 16);
        p.y = (u32)f2bf(v.z) | ((u32)f2bf(v.w) << 16);
        *(uint2*)(resB + m * D_DIM + d0 + jj4) = p;
    }
}

__global__ void k_padzero(float* resF, u16* resB) {
    size_t idx = (size_t)blockIdx.x * 256 + threadIdx.x;   // 65536 = 64*1024
    resF[(size_t)M_TOT * D_DIM + idx] = 0.0f;
    resB[(size_t)M_TOT * D_DIM + idx] = 0;
}

// --------------------------- fused WMMA distance GEMM + per-row argmin
#define BM 128
#define BN 64
#define BK 64
#define ASTR 144   // 128 data bytes + 16 pad
#define BSTR 144
#define BUFSZ (BM*ASTR + BN*BSTR)   // 27648 bytes per stage buffer

__global__ __launch_bounds__(256) void k_gemm_argmin(const u16* __restrict__ resB,
                                                     const u16* __restrict__ cbB,
                                                     const float* __restrict__ c2,
                                                     int K,
                                                     int* __restrict__ codes) {
    __shared__ __align__(16) unsigned char lds[2 * BUFSZ];

    int tid  = threadIdx.x;
    int lane = tid & 31, wave = tid >> 5;
    int half = lane >> 4, l16 = lane & 15;
    int m0 = blockIdx.x * BM;

    // cooperative async-load assignments
    int arow = tid >> 1, ahalf = tid & 1;     // 128 rows, 64B per half-row
    const u16* agp = resB + (size_t)(m0 + arow) * D_DIM + ahalf * 32;
    int brow = tid >> 2, bq = tid & 3;        // 64 rows, 32B quarters

    // All LDS addressing as u32 byte offsets (flat shared address keeps the
    // wave-relative LDS offset in its low 32 bits; same value feeds the async
    // VDST and the ds_load reads).
    u32 lds0 = (u32)(uintptr_t)(void*)lds;
    u32 aoff0 = lds0 + arow * ASTR + ahalf * 64;                 // async dst, A
    u32 boff0 = lds0 + BM * ASTR + brow * BSTR + bq * 32;        // async dst, B
    u32 afr0  = lds0 + (wave * 16 + l16) * ASTR + half * 32;     // A frag base
    u32 bfr0  = lds0 + BM * ASTR + l16 * BSTR + half * 32;       // B frag base

    float bestv[8];
    int   besti[8];
    #pragma unroll
    for (int j = 0; j < 8; j++) { bestv[j] = 3.4e38f; besti[j] = 0; }

    for (int n0 = 0; n0 < K; n0 += BN) {
        const u16* bgp = cbB + (size_t)(n0 + brow) * D_DIM + bq * 16;
        v8f acc[4];
        #pragma unroll
        for (int t = 0; t < 4; t++) acc[t] = (v8f){0,0,0,0,0,0,0,0};

        // prologue: fill buffer 0 with the first K-tile (6 async ops / wave)
        async_b128(aoff0, agp, 0);
        async_b128(aoff0, agp, 16);
        async_b128(aoff0, agp, 32);
        async_b128(aoff0, agp, 48);
        async_b128(boff0, bgp, 0);
        async_b128(boff0, bgp, 16);

        u32 pb = 0;   // current buffer byte offset (0 or BUFSZ)
        #pragma unroll
        for (int k0 = 0; k0 < D_DIM - BK; k0 += BK) {
            // issue next tile into the other buffer, overlapping compute
            u32 qb = pb ^ BUFSZ;
            const u16* ga = agp + (k0 + BK);
            const u16* gb = bgp + (k0 + BK);
            async_b128(aoff0 + qb, ga, 0);
            async_b128(aoff0 + qb, ga, 16);
            async_b128(aoff0 + qb, ga, 32);
            async_b128(aoff0 + qb, ga, 48);
            async_b128(boff0 + qb, gb, 0);
            async_b128(boff0 + qb, gb, 16);
            // async loads complete in order: <=6 left means current tile done
            asm volatile("s_wait_asynccnt 0x6" ::: "memory");
            __syncthreads();   // current buffer visible to all waves

            #pragma unroll
            for (int ks = 0; ks < 2; ks++) {
                union { u32x4 u[2]; v16bf v; } af;
                u32 ab = afr0 + pb + ks * 64;
                af.u[0] = lds_read16(ab);
                af.u[1] = lds_read16(ab + 16);
                union { u32x4 u[2]; v16bf v; } bf[4];
                #pragma unroll
                for (int t = 0; t < 4; t++) {
                    u32 bb = bfr0 + pb + t * (16 * BSTR) + ks * 64;
                    bf[t].u[0] = lds_read16(bb);
                    bf[t].u[1] = lds_read16(bb + 16);
                }
                #pragma unroll
                for (int t = 0; t < 4; t++) {
                    acc[t] = __builtin_amdgcn_wmma_f32_16x16x32_bf16(
                        false, af.v, false, bf[t].v, (short)0, acc[t], false, false);
                }
            }
            __syncthreads();   // reads done before this buffer is re-filled
            pb = qb;
        }
        // epilogue: last K-tile, nothing left in flight
        asm volatile("s_wait_asynccnt 0x0" ::: "memory");
        __syncthreads();
        #pragma unroll
        for (int ks = 0; ks < 2; ks++) {
            union { u32x4 u[2]; v16bf v; } af;
            u32 ab = afr0 + pb + ks * 64;
            af.u[0] = lds_read16(ab);
            af.u[1] = lds_read16(ab + 16);
            union { u32x4 u[2]; v16bf v; } bf[4];
            #pragma unroll
            for (int t = 0; t < 4; t++) {
                u32 bb = bfr0 + pb + t * (16 * BSTR) + ks * 64;
                bf[t].u[0] = lds_read16(bb);
                bf[t].u[1] = lds_read16(bb + 16);
            }
            #pragma unroll
            for (int t = 0; t < 4; t++) {
                acc[t] = __builtin_amdgcn_wmma_f32_16x16x32_bf16(
                    false, af.v, false, bf[t].v, (short)0, acc[t], false, false);
            }
        }
        __syncthreads();   // before next chunk's prologue refills buffer 0

        // score = ||c||^2 - 2 x.c ; running argmin (first-min tie-break)
        #pragma unroll
        for (int t = 0; t < 4; t++) {
            int n = n0 + t * 16 + l16;
            float cc = c2[n];
            #pragma unroll
            for (int j = 0; j < 8; j++) {
                float sc = cc - 2.0f * acc[t][j];
                if (sc < bestv[j] || (sc == bestv[j] && n < besti[j])) {
                    bestv[j] = sc; besti[j] = n;
                }
            }
        }
    }
    // reduce over 16 lanes within each half-wave (rows j / j+8)
    #pragma unroll
    for (int m = 1; m <= 8; m <<= 1) {
        #pragma unroll
        for (int j = 0; j < 8; j++) {
            float ov = __shfl_xor(bestv[j], m, 32);
            int   oi = __shfl_xor(besti[j], m, 32);
            if (ov < bestv[j] || (ov == bestv[j] && oi < besti[j])) {
                bestv[j] = ov; besti[j] = oi;
            }
        }
    }
    int rowbase = m0 + wave * 16 + half * 8;
    if (l16 == 0) {
        #pragma unroll
        for (int j = 0; j < 8; j++) {
            int r = rowbase + j;
            if (r < M_TOT) codes[r] = besti[j];
        }
    }
}

// ------------------ residual -= cb[idx]; refresh bf16 shadow; sum-squares
__global__ __launch_bounds__(256) void k_update(float* __restrict__ resF,
                                                u16* __restrict__ resB,
                                                const float* __restrict__ cb,
                                                const int* __restrict__ codes,
                                                float* __restrict__ ssq) {
    int m = blockIdx.x;
    int idx = codes[m];
    int tid = threadIdx.x;
    float4 rv = ((const float4*)(resF + (size_t)m * D_DIM))[tid];
    float4 cv = ((const float4*)(cb + (size_t)idx * D_DIM))[tid];
    rv.x -= cv.x; rv.y -= cv.y; rv.z -= cv.z; rv.w -= cv.w;
    ((float4*)(resF + (size_t)m * D_DIM))[tid] = rv;
    uint2 p;
    p.x = (u32)f2bf(rv.x) | ((u32)f2bf(rv.y) << 16);
    p.y = (u32)f2bf(rv.z) | ((u32)f2bf(rv.w) << 16);
    ((uint2*)(resB + (size_t)m * D_DIM))[tid] = p;
    float s = rv.x*rv.x + rv.y*rv.y + rv.z*rv.z + rv.w*rv.w;
    #pragma unroll
    for (int mm = 16; mm >= 1; mm >>= 1) s += __shfl_xor(s, mm, 32);
    __shared__ float part[8];
    int lane = tid & 31, wave = tid >> 5;
    if (lane == 0) part[wave] = s;
    __syncthreads();
    if (tid == 0) {
        float t = 0.f;
        #pragma unroll
        for (int i = 0; i < 8; i++) t += part[i];
        atomicAdd(ssq, t);
    }
}

// ---------------------------------------------------------- emit codes
__global__ void k_codes(const int* __restrict__ codes, float* __restrict__ out) {
    int i = blockIdx.x * 256 + threadIdx.x;    // (b, s, t) flat, 192000 total
    int t  = i % T_SZ;
    int bs = i / T_SZ;
    int s = bs % NQ, b = bs / NQ;
    out[i] = (float)codes[(size_t)s * M_TOT + (size_t)b * T_SZ + t];
}

// ------------------------------------- out[b,d,t] = z[b,d,t] - res[b*T+t, d]
__global__ __launch_bounds__(256) void k_out(const float* __restrict__ z,
                                             const float* __restrict__ resF,
                                             float* __restrict__ out) {
    __shared__ float tile[32][33];             // [d_local][t_local]
    int b = blockIdx.x, t0 = blockIdx.y * 32, d0 = blockIdx.z * 32;
    int tid = threadIdx.x;
    int i  = tid >> 3;          // t-local row
    int j4 = (tid & 7) * 4;     // d-local
    int t = t0 + i;
    if (t < T_SZ) {
        float4 v = *(const float4*)(resF + ((size_t)b * T_SZ + t) * D_DIM + d0 + j4);
        tile[j4 + 0][i] = v.x;
        tile[j4 + 1][i] = v.y;
        tile[j4 + 2][i] = v.z;
        tile[j4 + 3][i] = v.w;
    }
    __syncthreads();
    int jj  = tid >> 3;         // d-local row
    int ii4 = (tid & 7) * 4;    // t-local
    size_t base = ((size_t)b * D_DIM + d0 + jj) * T_SZ;
    #pragma unroll
    for (int e = 0; e < 4; e++) {
        int t2 = t0 + ii4 + e;
        if (t2 < T_SZ) out[base + t2] = z[base + t2] - tile[jj][ii4 + e];
    }
}

// ------------------------------------------------------------------ loss
__global__ void k_loss(const float* __restrict__ ssq, float* __restrict__ out) {
    if (threadIdx.x == 0) {
        float tot = 0.f;
        for (int s = 0; s < NQ; s++) tot += ssq[s];
        // per-stage loss = 1.25 * ssq_s / (M*D); total = mean over 8 stages
        out[0] = tot * 1.25f / (8.0f * (float)M_TOT * (float)D_DIM);
    }
}

extern "C" void kernel_launch(void* const* d_in, const int* in_sizes, int n_in,
                              void* d_out, int out_size, void* d_ws, size_t ws_size,
                              hipStream_t stream) {
    const float* z   = (const float*)d_in[0];   // [16,1024,1500]
    const float* cb0 = (const float*)d_in[1];   // [2048,1024]
    const float* cbr = (const float*)d_in[2];   // [7,1024,1024]
    float* out = (float*)d_out;

    unsigned char* ws = (unsigned char*)d_ws;
    size_t off = 0;
    float* resF = (float*)(ws + off); off += (size_t)M_PAD * D_DIM * 4;   // 98.6 MB
    u16*   resB = (u16*)  (ws + off); off += (size_t)M_PAD * D_DIM * 2;   // 49.3 MB
    u16*   cbB  = (u16*)  (ws + off); off += (size_t)NCB * D_DIM * 2;     // 18.9 MB
    float* c2   = (float*)(ws + off); off += (size_t)NCB * 4;
    int*   codes= (int*)  (ws + off); off += (size_t)NQ * M_TOT * 4;
    float* ssq  = (float*)(ws + off); off += 256;

    k_zero<<<1, 32, 0, stream>>>(ssq);
    k_prep_cb<<<NCB, 256, 0, stream>>>(cb0, cbr, cbB, c2);
    dim3 tg(B_SZ, (T_SZ + 31) / 32, D_DIM / 32);   // 16 x 47 x 32
    k_transpose<<<tg, 256, 0, stream>>>(z, resF, resB);
    k_padzero<<<256, 256, 0, stream>>>(resF, resB);

    for (int s = 0; s < NQ; s++) {
        int K = (s == 0) ? KSTAGE0 : KREST;
        size_t cbrow = (s == 0) ? 0 : (size_t)(KSTAGE0 + (s - 1) * KREST);
        const float* cbf = (s == 0) ? cb0 : cbr + (size_t)(s - 1) * KREST * D_DIM;
        k_gemm_argmin<<<M_PAD / BM, 256, 0, stream>>>(
            resB, cbB + cbrow * D_DIM, c2 + cbrow, K, codes + (size_t)s * M_TOT);
        k_update<<<M_TOT, 256, 0, stream>>>(
            resF, resB, cbf, codes + (size_t)s * M_TOT, ssq + s);
    }

    k_codes<<<(int)(CODE_ELEMS / 256), 256, 0, stream>>>(codes, out + OUT_ELEMS);
    k_out<<<tg, 256, 0, stream>>>(z, resF, out);
    k_loss<<<1, 32, 0, stream>>>(ssq, out + OUT_ELEMS + CODE_ELEMS);
}
